// ThreeBodyInteractions_74998718922917
// MI455X (gfx1250) — matrix operand order repaired and checked by reading, not compiled, not run on hardware
//
#include <hip/hip_runtime.h>
#include <math.h>

typedef __attribute__((ext_vector_type(16))) _Float16 v16h;
typedef __attribute__((ext_vector_type(8)))  float    v8f;
typedef __attribute__((ext_vector_type(4)))  int      v4i;

#define N_ATOMS   5000
#define N_EDGES   200000
#define N_TRIPLES 4000000
#define D_NODE    128
#define D_EDGE    128
#define N_BASIS   9

#define ATOM_STRIDE 12     // atoms padded row stride (16B-aligned float4 gathers)
#define NB_STRIDE   16     // new_bonds padded row stride (zeros in cols 9..15)

#if __has_builtin(__builtin_amdgcn_global_load_async_to_lds_b128)
#define HAVE_ASYNC_LDS 1
#else
#define HAVE_ASYNC_LDS 0
#endif

__device__ __forceinline__ void wait_asynccnt0() {
#if __has_builtin(__builtin_amdgcn_s_wait_asynccnt)
    __builtin_amdgcn_s_wait_asynccnt(0);
#else
    asm volatile("s_wait_asynccnt 0x0" ::: "memory");
#endif
}

__device__ __forceinline__ float sigmoidf_fast(float x) {
    return 1.0f / (1.0f + __expf(-x));
}

// ---------------------------------------------------------------------------
// Kernel 0: zero the padded segment-sum accumulator (ws is poisoned 0xAA and
// timing replays reuse it). The stride-16 padding zeros double as the K=9..15
// zero-fill of the edge kernel's WMMA A-fragment.
// ---------------------------------------------------------------------------
__global__ void zero_ws_kernel(float* __restrict__ p, long long n) {
    long long i = (long long)blockIdx.x * blockDim.x + threadIdx.x;
    long long stride = (long long)gridDim.x * blockDim.x;
    for (; i < n; i += stride) p[i] = 0.0f;
}

// ---------------------------------------------------------------------------
// Kernel P: pre-swizzle the three weight matrices into WMMA B-fragment order
// (f16, zero-padded). Fragment layout: frag[chunk][lane][h], 16 f16 (32 B)
// per lane -> the GEMM kernels load each B fragment with one aligned vector
// load instead of 16 predicated scalar loads.
//   B (32x16 KxN) per-lane mapping: N = lane%16, K = 8*(lane>=16) + (h&7) + 16*(h>>3).
// ---------------------------------------------------------------------------
__global__ void __launch_bounds__(256)
prep_frags_kernel(const float* __restrict__ W_atom,
                  const float* __restrict__ W_gate,
                  const float* __restrict__ W_sig,
                  _Float16* __restrict__ fragA,   // [4 kchunks][32][16]
                  _Float16* __restrict__ fragG,   // [8 ntiles ][32][16]
                  _Float16* __restrict__ fragS) { // [8 ntiles ][32][16]
    for (int e = threadIdx.x; e < 4 * 32 * 16; e += 256) {
        const int h = e & 15, lane = (e >> 4) & 31, kc = e >> 9;
        const int lm = lane & 15, grp = lane >> 4;
        const int k = kc * 32 + grp * 8 + (h & 7) + (h >> 3) * 16; // 0..127
        fragA[e] = (lm < N_BASIS) ? (_Float16)W_atom[k * N_BASIS + lm]
                                  : (_Float16)0.0f;
    }
    for (int e = threadIdx.x; e < 8 * 32 * 16; e += 256) {
        const int h = e & 15, lane = (e >> 4) & 31, nt = e >> 9;
        const int lm = lane & 15, grp = lane >> 4;
        const int k = grp * 8 + (h & 7) + (h >> 3) * 16;      // K padded 9->32
        const int col = nt * 16 + lm;
        const bool ok = (k < N_BASIS);
        fragG[e] = ok ? (_Float16)W_gate[k * D_EDGE + col] : (_Float16)0.0f;
        fragS[e] = ok ? (_Float16)W_sig [k * D_EDGE + col] : (_Float16)0.0f;
    }
}

// ---------------------------------------------------------------------------
// Kernel 1: atoms = sigmoid(node_feat @ W_atom + b_atom), padded to stride 12.
// One wave per 16-row tile, K=128 as 4 WMMA chunks. A-fragment per lane is
// two contiguous 8-float runs (K = grp*8+{0..7} and 16+grp*8+{0..7}) -> four
// unpredicated global_load_b128 (row clamped; store masked instead).
// ---------------------------------------------------------------------------
__global__ void __launch_bounds__(32)
atoms_kernel(const float* __restrict__ node_feat,
             const _Float16* __restrict__ fragA,
             const float* __restrict__ b_atom,
             float* __restrict__ atoms12) {
    const int tile = blockIdx.x;
    const int lane = threadIdx.x;
    const int grp  = lane >> 4;
    const int lm   = lane & 15;
    int row = tile * 16 + lm;
    if (row > N_ATOMS - 1) row = N_ATOMS - 1;   // clamp: no exec diamonds on loads

    v8f c = {};
    #pragma unroll
    for (int kc = 0; kc < 4; ++kc) {
        const float* base = node_feat + row * D_NODE + kc * 32 + grp * 8;
        const float4 x0 = *(const float4*)(base + 0);
        const float4 x1 = *(const float4*)(base + 4);
        const float4 y0 = *(const float4*)(base + 16);
        const float4 y1 = *(const float4*)(base + 20);
        v16h a;
        a[0] = (_Float16)x0.x; a[1] = (_Float16)x0.y;
        a[2] = (_Float16)x0.z; a[3] = (_Float16)x0.w;
        a[4] = (_Float16)x1.x; a[5] = (_Float16)x1.y;
        a[6] = (_Float16)x1.z; a[7] = (_Float16)x1.w;
        a[8]  = (_Float16)y0.x; a[9]  = (_Float16)y0.y;
        a[10] = (_Float16)y0.z; a[11] = (_Float16)y0.w;
        a[12] = (_Float16)y1.x; a[13] = (_Float16)y1.y;
        a[14] = (_Float16)y1.z; a[15] = (_Float16)y1.w;
        const v16h b = *(const v16h*)(fragA + (kc * 32 + lane) * 16); // 32B aligned
        c = __builtin_amdgcn_wmma_f32_16x16x32_f16(
                false, a, false, b, (short)0, c, false, false);
    }

    if (lm < N_BASIS) {                          // after WMMA: divergence is fine
        const float bias = b_atom[lm];
        #pragma unroll
        for (int r = 0; r < 8; ++r) {
            const int m = tile * 16 + r + grp * 8;   // D: M = r + 8*grp, N = lm
            if (m < N_ATOMS)
                atoms12[m * ATOM_STRIDE + lm] = sigmoidf_fast(c[r] + bias);
        }
    }
}

// ---------------------------------------------------------------------------
// Kernel 2: segment-sum of 4M sorted triples.
// Each block stages 1024 triples (basis 36KB + ids 8KB) into LDS. Full blocks
// use CDNA5 async global->LDS copies (no VGPR round-trip, ASYNCcnt-tracked);
// the single tail block uses guarded synchronous staging. Then each thread
// run-length-accumulates 8 consecutive triples and flushes with atomics only
// at segment boundaries. atoms12 (240KB) and graph_dst (800KB) stay resident
// in the 192MB L2.
// ---------------------------------------------------------------------------
#define SC_BLOCK 128
#define SC_TPT   8
#define SC_TPB   (SC_BLOCK * SC_TPT)   // 1024 triples per block

__global__ void __launch_bounds__(SC_BLOCK)
scatter_kernel(const float* __restrict__ three_basis,
               const float* __restrict__ atoms12,
               const int*   __restrict__ graph_dst,
               const int*   __restrict__ lg_dst,
               const int*   __restrict__ segment_ids,
               float* __restrict__ new_bonds16) {
    __shared__ float s_basis[SC_TPB * N_BASIS];  // 36,864 B
    __shared__ int   s_seg[SC_TPB];              //  4,096 B
    __shared__ int   s_lg [SC_TPB];              //  4,096 B

    const long long blockBase = (long long)blockIdx.x * SC_TPB;
    const int tid = threadIdx.x;

    const long long gbase = blockBase * N_BASIS;          // 16B-aligned
    if (blockBase + SC_TPB <= N_TRIPLES) {                // full block: no guards
#if HAVE_ASYNC_LDS
        // ---- async DMA: global_load_async_to_lds_b128, EXEC all-ones ----
        #pragma unroll
        for (int f = 0; f < SC_TPB * N_BASIS / 4; f += SC_BLOCK) {
            const int fi = f + tid;
            __builtin_amdgcn_global_load_async_to_lds_b128(
                (v4i*)(three_basis + gbase + (long long)fi * 4),
                (v4i*)(s_basis + fi * 4),
                /*offset=*/0, /*cpol=*/0);
        }
#else
        #pragma unroll
        for (int f = 0; f < SC_TPB * N_BASIS / 4; f += SC_BLOCK) {
            const int fi = f + tid;
            *(float4*)(s_basis + fi * 4) =
                *(const float4*)(three_basis + gbase + (long long)fi * 4);
        }
#endif
        #pragma unroll
        for (int k = 0; k < SC_TPB; k += SC_BLOCK) {
            s_seg[k + tid] = segment_ids[blockBase + k + tid];
            s_lg [k + tid] = lg_dst     [blockBase + k + tid];
        }
#if HAVE_ASYNC_LDS
        wait_asynccnt0();      // own async copies landed in LDS before barrier
#endif
    } else {                                              // tail block (sync path)
        const long long totalF = (long long)N_TRIPLES * N_BASIS;
        for (int f = tid; f < SC_TPB * N_BASIS / 4; f += SC_BLOCK) {
            const long long g = gbase + (long long)f * 4;
            float4 v = make_float4(0.f, 0.f, 0.f, 0.f);
            if (g + 3 < totalF) v = *(const float4*)(three_basis + g);
            *(float4*)(s_basis + f * 4) = v;
        }
        for (int k = tid; k < SC_TPB; k += SC_BLOCK) {
            const long long t = blockBase + k;
            s_seg[k] = (t < N_TRIPLES) ? segment_ids[t] : -1;
            s_lg [k] = (t < N_TRIPLES) ? lg_dst[t] : 0;
        }
    }
    __syncthreads();

    // ---- run-length accumulate over sorted segment ids ----
    const int tbase = tid * SC_TPT;
    float acc[N_BASIS];
    int cur = -1;
    for (int i = 0; i < SC_TPT; ++i) {
        const int li = tbase + i;
        if (blockBase + li >= N_TRIPLES) break;
        const int seg = s_seg[li];
        if (seg != cur) {
            if (cur >= 0) {
                float* dst = new_bonds16 + (long long)cur * NB_STRIDE;
                #pragma unroll
                for (int j = 0; j < N_BASIS; ++j) atomicAdd(&dst[j], acc[j]);
            }
            cur = seg;
            #pragma unroll
            for (int j = 0; j < N_BASIS; ++j) acc[j] = 0.0f;
        }
        const int atom = graph_dst[s_lg[li]];
        const float* __restrict__ ap = atoms12 + atom * ATOM_STRIDE;
        const float4 a0 = *(const float4*)(ap + 0);      // 16B-aligned gathers
        const float4 a1 = *(const float4*)(ap + 4);
        const float4 a2 = *(const float4*)(ap + 8);
        const float* bp = s_basis + li * N_BASIS;        // ds loads
        acc[0] = fmaf(bp[0], a0.x, acc[0]);
        acc[1] = fmaf(bp[1], a0.y, acc[1]);
        acc[2] = fmaf(bp[2], a0.z, acc[2]);
        acc[3] = fmaf(bp[3], a0.w, acc[3]);
        acc[4] = fmaf(bp[4], a1.x, acc[4]);
        acc[5] = fmaf(bp[5], a1.y, acc[5]);
        acc[6] = fmaf(bp[6], a1.z, acc[6]);
        acc[7] = fmaf(bp[7], a1.w, acc[7]);
        acc[8] = fmaf(bp[8], a2.x, acc[8]);
    }
    if (cur >= 0) {
        float* dst = new_bonds16 + (long long)cur * NB_STRIDE;
        #pragma unroll
        for (int j = 0; j < N_BASIS; ++j) atomicAdd(&dst[j], acc[j]);
    }
}

// ---------------------------------------------------------------------------
// Kernel 3: out = edge_feat + silu(NB @ W_gate + bg) * sigmoid(NB @ W_sig + bs)
// One wave per 16-edge tile, 16 WMMAs per wave. A-fragment = two aligned
// b128 loads from the zero-padded stride-16 accumulator (no masking needed);
// B-fragments come pre-swizzled from workspace.
// ---------------------------------------------------------------------------
__global__ void __launch_bounds__(256)
edge_kernel(const float* __restrict__ edge_feat,
            const float* __restrict__ new_bonds16,
            const _Float16* __restrict__ fragG,
            const _Float16* __restrict__ fragS,
            const float* __restrict__ b_gate,
            const float* __restrict__ b_sig,
            float* __restrict__ out) {
    const int wave = (blockIdx.x * blockDim.x + threadIdx.x) >> 5;
    if (wave >= N_EDGES / 16) return;            // wave-uniform exit
    const int lane = threadIdx.x & 31;
    const int grp  = lane >> 4;
    const int lm   = lane & 15;
    const int row  = wave * 16 + lm;

    // A: K = grp*8 + {0..7} live (cols 9..15 are stored zeros); K>=16 -> 0
    v16h a;
    {
        const float* pn = new_bonds16 + row * NB_STRIDE + grp * 8;
        const float4 x0 = *(const float4*)(pn + 0);
        const float4 x1 = *(const float4*)(pn + 4);
        a[0] = (_Float16)x0.x; a[1] = (_Float16)x0.y;
        a[2] = (_Float16)x0.z; a[3] = (_Float16)x0.w;
        a[4] = (_Float16)x1.x; a[5] = (_Float16)x1.y;
        a[6] = (_Float16)x1.z; a[7] = (_Float16)x1.w;
        #pragma unroll
        for (int h = 8; h < 16; ++h) a[h] = (_Float16)0.0f;
    }

    #pragma unroll
    for (int nt = 0; nt < D_EDGE / 16; ++nt) {
        const int col = nt * 16 + lm;
        const v16h bg = *(const v16h*)(fragG + (nt * 32 + lane) * 16);
        const v16h bs = *(const v16h*)(fragS + (nt * 32 + lane) * 16);
        v8f cg = {}, cs = {};
        cg = __builtin_amdgcn_wmma_f32_16x16x32_f16(
                false, a, false, bg, (short)0, cg, false, false);
        cs = __builtin_amdgcn_wmma_f32_16x16x32_f16(
                false, a, false, bs, (short)0, cs, false, false);

        const float biasg = b_gate[col];
        const float biass = b_sig[col];
        #pragma unroll
        for (int r = 0; r < 8; ++r) {
            const int m = wave * 16 + r + grp * 8;   // D: M = r + 8*grp, N = lm
            const float g = cg[r] + biasg;
            const float s = cs[r] + biass;
            const float upd = (g * sigmoidf_fast(g)) * sigmoidf_fast(s);
            const int idx = m * D_EDGE + col;        // 64B-coalesced per half-wave
            out[idx] = edge_feat[idx] + upd;
        }
    }
}

// ---------------------------------------------------------------------------
extern "C" void kernel_launch(void* const* d_in, const int* in_sizes, int n_in,
                              void* d_out, int out_size, void* d_ws, size_t ws_size,
                              hipStream_t stream) {
    const float* node_feat   = (const float*)d_in[0];
    const float* edge_feat   = (const float*)d_in[1];
    const float* three_basis = (const float*)d_in[2];
    // d_in[3] three_cutoff : dead code in reference
    const float* W_atom      = (const float*)d_in[4];
    const float* b_atom      = (const float*)d_in[5];
    const float* W_gate      = (const float*)d_in[6];
    const float* b_gate      = (const float*)d_in[7];
    const float* W_sig       = (const float*)d_in[8];
    const float* b_sig       = (const float*)d_in[9];
    const int*   graph_dst   = (const int*)d_in[10];
    // d_in[11] lg_src : dead code in reference
    const int*   lg_dst      = (const int*)d_in[12];
    const int*   segment_ids = (const int*)d_in[13];
    float*       out         = (float*)d_out;

    // Workspace layout (all 256B-aligned):
    //   atoms12     : 5000*12  f32 = 240,000 B
    //   new_bonds16 : 200000*16 f32 = 12,800,000 B
    //   fragA       : 4*32*16 f16 =  4,096 B
    //   fragG,fragS : 8*32*16 f16 =  8,192 B each
    char* ws = (char*)d_ws;
    float*    atoms12     = (float*)ws;
    float*    new_bonds16 = (float*)(ws + 240128);
    _Float16* fragA       = (_Float16*)(ws + 240128 + 12800000);
    _Float16* fragG       = (_Float16*)(ws + 240128 + 12800000 + 4096);
    _Float16* fragS       = (_Float16*)(ws + 240128 + 12800000 + 4096 + 8192);

    // 0) zero the padded accumulator
    zero_ws_kernel<<<1024, 256, 0, stream>>>(new_bonds16,
                                             (long long)N_EDGES * NB_STRIDE);

    // P) pre-swizzle weights into WMMA B-fragment order
    prep_frags_kernel<<<1, 256, 0, stream>>>(W_atom, W_gate, W_sig,
                                             fragA, fragG, fragS);

    // 1) atoms = sigmoid(node_feat @ W_atom + b_atom)   (WMMA)
    atoms_kernel<<<(N_ATOMS + 15) / 16, 32, 0, stream>>>(node_feat, fragA,
                                                         b_atom, atoms12);

    // 2) segment-sum of 4M triples (async-LDS staged, run-length + atomics)
    scatter_kernel<<<(N_TRIPLES + SC_TPB - 1) / SC_TPB, SC_BLOCK, 0, stream>>>(
        three_basis, atoms12, graph_dst, lg_dst, segment_ids, new_bonds16);

    // 3) gated-MLP edge update + residual   (WMMA)
    edge_kernel<<<((N_EDGES / 16) * 32 + 255) / 256, 256, 0, stream>>>(
        edge_feat, new_bonds16, fragG, fragS, b_gate, b_sig, out);
}
